// HawkeyeMoE_33500744909265
// MI455X (gfx1250) — compile-verified, matrix operand order checked by reading, not run on hardware
//
#include <hip/hip_runtime.h>

// ---------------------------------------------------------------------------
// HawkeyeMoE for MI455X (gfx1250): all GEMMs via v_wmma_f32_16x16x32_f16,
// wave32, 8 waves/block, 128x128x32 tiles, double-buffered LDS pipeline.
// ---------------------------------------------------------------------------

typedef _Float16 v16h __attribute__((ext_vector_type(16)));
typedef float    v8f  __attribute__((ext_vector_type(8)));
typedef float    f4v  __attribute__((ext_vector_type(4)));
typedef _Float16 h8v  __attribute__((ext_vector_type(8)));
typedef _Float16 h2v  __attribute__((ext_vector_type(2)));

union AFrag { v16h v; h8v h[2]; _Float16 e[16]; };
union CFrag { v8f  v; float e[8]; };

__device__ __forceinline__ void hk_load8(const float* __restrict__ p, float* r) {
  const f4v* q = (const f4v*)p;
  const f4v a = q[0], b = q[1];
#pragma unroll
  for (int j = 0; j < 4; j++) { r[j] = a[j]; r[4 + j] = b[j]; }
}
__device__ __forceinline__ void hk_load8(const _Float16* __restrict__ p, float* r) {
  const h8v a = *(const h8v*)p;
#pragma unroll
  for (int j = 0; j < 8; j++) r[j] = (float)a[j];
}

#define GBM 128
#define GBN 128
#define GBK 32

// C[M,N] = alpha * A(f16)[M,K] @ op(B)[K,N] + bias + residual, batched over z.
// Assumes K % 32 == 0 and N % 8 == 0 (true for every GEMM in this model).
template<typename BT, typename CT, bool TRANSB>
__global__ __launch_bounds__(256)
void hawkeye_gemm_wmma(const _Float16* __restrict__ A, int lda, long a_s0, long a_s1,
                       const BT* __restrict__ Bm, int ldb, long b_s0, long b_s1,
                       CT* __restrict__ C, int ldc, long c_s0, long c_s1,
                       const float* __restrict__ bias,
                       const float* __restrict__ res,
                       float alpha, int M, int N, int K, int zdiv) {
  __shared__ __align__(16) _Float16 As[2][GBM * GBK];
  __shared__ __align__(16) _Float16 Bs[2][GBN * GBK];   // transposed: [n][k]

  const int z = blockIdx.z;
  const long z0 = z / zdiv, z1 = z % zdiv;
  A  += z0 * a_s0 + z1 * a_s1;
  Bm += z0 * b_s0 + z1 * b_s1;
  C  += z0 * c_s0 + z1 * c_s1;
  const float* R = res ? (res + z0 * c_s0 + z1 * c_s1) : nullptr;

  const int row0 = blockIdx.y * GBM;
  const int col0 = blockIdx.x * GBN;
  const int tid  = threadIdx.x;
  const int wave = tid >> 5;
  const int lane = tid & 31;
  const int wm   = (wave >> 1) * 32;        // 8 waves: 4 x 2 grid of 32x64
  const int wn   = (wave & 1) * 64;
  const int lrow = lane & 15;
  const int lkb  = (lane >> 4) * 8;

  // global->reg staging assignments (256 threads)
  const int arow = tid >> 1;                 // A: row 0..127
  const int acol = (tid & 1) * 16;           // A: 16 contiguous halves
  const int bkk  = (tid >> 4) * 2;           // B (no-trans): k pair 0..30
  const int bno  = (tid & 15) * 8;           // B (no-trans): 8 contiguous cols
  const int bn   = tid >> 1;                 // B (trans): row of B^T
  const int bcc  = (tid & 1) * 16;

  CFrag acc[2][4];
#pragma unroll
  for (int tm = 0; tm < 2; tm++)
#pragma unroll
    for (int tn = 0; tn < 4; tn++)
#pragma unroll
      for (int j = 0; j < 8; j++) acc[tm][tn].e[j] = 0.f;

  h8v ra0, ra1;
  float rb[16];

  auto gload = [&](int k0) {
    // ---- A: one row, 16 contiguous halves (2 x b128)
    if (row0 + arow < M) {
      const h8v* p = (const h8v*)(A + (long)(row0 + arow) * lda + k0 + acol);
      ra0 = p[0]; ra1 = p[1];
    } else {
#pragma unroll
      for (int j = 0; j < 8; j++) { ra0[j] = (_Float16)0.f; ra1[j] = (_Float16)0.f; }
    }
    // ---- B
    if (TRANSB) {                            // B stored [N,K], contiguous in k
      if (col0 + bn < N) {
        hk_load8(Bm + (long)(col0 + bn) * ldb + k0 + bcc, rb);
        hk_load8(Bm + (long)(col0 + bn) * ldb + k0 + bcc + 8, rb + 8);
      } else {
#pragma unroll
        for (int j = 0; j < 16; j++) rb[j] = 0.f;
      }
    } else {                                 // B stored [K,N], contiguous in n
      if (col0 + bno < N) {
        hk_load8(Bm + (long)(k0 + bkk) * ldb + col0 + bno, rb);
        hk_load8(Bm + (long)(k0 + bkk + 1) * ldb + col0 + bno, rb + 8);
      } else {
#pragma unroll
        for (int j = 0; j < 16; j++) rb[j] = 0.f;
      }
    }
  };

  auto sstore = [&](int buf) {
    *(h8v*)&As[buf][arow * GBK + acol]     = ra0;
    *(h8v*)&As[buf][arow * GBK + acol + 8] = ra1;
    if (TRANSB) {
      h8v b0, b1;
#pragma unroll
      for (int j = 0; j < 8; j++) { b0[j] = (_Float16)rb[j]; b1[j] = (_Float16)rb[8 + j]; }
      *(h8v*)&Bs[buf][bn * GBK + bcc]     = b0;
      *(h8v*)&Bs[buf][bn * GBK + bcc + 8] = b1;
    } else {
#pragma unroll
      for (int j = 0; j < 8; j++) {          // packed (k, k+1) pair -> b32 store
        h2v pv;
        pv[0] = (_Float16)rb[j];
        pv[1] = (_Float16)rb[8 + j];
        *(h2v*)&Bs[buf][(bno + j) * GBK + bkk] = pv;
      }
    }
  };

  auto compute = [&](int buf) {
    AFrag af[2], bf[4];
#pragma unroll
    for (int t = 0; t < 2; t++) {
      const int fr = wm + t * 16 + lrow;
      af[t].h[0] = *(const h8v*)&As[buf][fr * GBK + lkb];
      af[t].h[1] = *(const h8v*)&As[buf][fr * GBK + lkb + 16];
    }
#pragma unroll
    for (int t = 0; t < 4; t++) {
      const int fn = wn + t * 16 + lrow;
      bf[t].h[0] = *(const h8v*)&Bs[buf][fn * GBK + lkb];
      bf[t].h[1] = *(const h8v*)&Bs[buf][fn * GBK + lkb + 16];
    }
#pragma unroll
    for (int tm = 0; tm < 2; tm++)
#pragma unroll
      for (int tn = 0; tn < 4; tn++)
        acc[tm][tn].v = __builtin_amdgcn_wmma_f32_16x16x32_f16(
            false, af[tm].v, false, bf[tn].v, (short)0, acc[tm][tn].v, false, false);
  };

  // software pipeline: load k+1 tile from global while WMMAs chew on tile k
  gload(0);
  sstore(0);
  __syncthreads();
  int cur = 0;
  for (int k0 = 0; k0 < K; k0 += GBK) {
    const bool more = (k0 + GBK) < K;
    if (more) gload(k0 + GBK);
    compute(cur);
    if (more) {
      sstore(cur ^ 1);
      __syncthreads();
      cur ^= 1;
    }
  }

  // epilogue: D layout: VGPR v -> M = v + (lane>=16 ? 8 : 0), N = lane & 15
#pragma unroll
  for (int tm = 0; tm < 2; tm++) {
#pragma unroll
    for (int tn = 0; tn < 4; tn++) {
      const int cc = col0 + wn + tn * 16 + lrow;
      if (cc < N) {
        const int rbase = row0 + wm + tm * 16 + (lane >> 4) * 8;
        const float bv = bias ? bias[cc] : 0.f;
#pragma unroll
        for (int vv = 0; vv < 8; vv++) {
          const int rr = rbase + vv;
          if (rr < M) {
            float val = acc[tm][tn].e[vv] * alpha + bv;
            if (R) val += R[(long)rr * ldc + cc];
            C[(long)rr * ldc + cc] = (CT)val;
          }
        }
      }
    }
  }
}

// ---------------------------------------------------------------------------
// Support kernels
// ---------------------------------------------------------------------------

__global__ void hawkeye_concat(const float* __restrict__ pose,
                               const float* __restrict__ scene,
                               float* __restrict__ x, int total) {
  const int idx = blockIdx.x * blockDim.x + threadIdx.x;
  if (idx >= total) return;
  const int c = idx & 1023;
  const int t = (idx >> 10) & 511;
  const int b = idx >> 19;
  x[idx] = (t < 256) ? pose[((long)b * 256 + t) * 1024 + c]
                     : scene[((long)b * 256 + (t - 256)) * 1024 + c];
}

__global__ void hawkeye_router(const float* __restrict__ x,
                               const float* __restrict__ w1, const float* __restrict__ b1,
                               const float* __restrict__ w2, const float* __restrict__ b2,
                               float* __restrict__ r) {
  const int tok = blockIdx.x;
  const int tid = threadIdx.x;
  __shared__ float red[4][256];
  const float* xr = x + (long)tok * 1024;
  float acc[4] = {0.f, 0.f, 0.f, 0.f};
  for (int j = tid; j < 1024; j += 256) {
    const float xv = xr[j];
#pragma unroll
    for (int e = 0; e < 4; e++) acc[e] += xv * w1[j * 4 + e];
  }
#pragma unroll
  for (int e = 0; e < 4; e++) red[e][tid] = acc[e];
  __syncthreads();
  for (int st = 128; st > 0; st >>= 1) {
    if (tid < st)
#pragma unroll
      for (int e = 0; e < 4; e++) red[e][tid] += red[e][tid + st];
    __syncthreads();
  }
  if (tid == 0) {
    float hd[4], rr[4], sum = 0.f;
#pragma unroll
    for (int e = 0; e < 4; e++) {
      const float zz = red[e][0] + b1[e];
      hd[e] = 0.5f * zz * (1.f + erff(zz * 0.70710678118654752f));  // exact gelu
    }
#pragma unroll
    for (int eo = 0; eo < 4; eo++) {
      float zz = b2[eo];
#pragma unroll
      for (int ei = 0; ei < 4; ei++) zz += hd[ei] * w2[ei * 4 + eo];
      rr[eo] = 1.f / (1.f + expf(-zz));
      sum += rr[eo];
    }
    sum = fmaxf(sum, 1e-8f);
#pragma unroll
    for (int e = 0; e < 4; e++) r[(long)tok * 4 + e] = rr[e] / sum;
  }
}

__global__ void hawkeye_rmsnorm(const float* __restrict__ h,
                                const float* __restrict__ w,
                                _Float16* __restrict__ out) {
  const int tok = blockIdx.x;
  const int tid = threadIdx.x;
  __shared__ float red[256];
  const float* row = h + (long)tok * 1024;
  float vals[4], ss = 0.f;
#pragma unroll
  for (int i = 0; i < 4; i++) {
    vals[i] = row[tid + i * 256];
    ss += vals[i] * vals[i];
  }
  red[tid] = ss;
  __syncthreads();
  for (int st = 128; st > 0; st >>= 1) {
    if (tid < st) red[tid] += red[tid + st];
    __syncthreads();
  }
  const float scale = rsqrtf(red[0] * (1.f / 1024.f) + 1e-5f);
#pragma unroll
  for (int i = 0; i < 4; i++)
    out[(long)tok * 1024 + tid + i * 256] =
        (_Float16)(vals[i] * scale * w[tid + i * 256]);
}

__global__ void hawkeye_softmax(const float* __restrict__ S,
                                _Float16* __restrict__ P) {
  const long row = blockIdx.x;
  const int tid = threadIdx.x;
  __shared__ float red[256];
  const float* sr = S + row * 512;
  const float v0 = sr[tid], v1 = sr[tid + 256];
  red[tid] = fmaxf(v0, v1);
  __syncthreads();
  for (int st = 128; st > 0; st >>= 1) {
    if (tid < st) red[tid] = fmaxf(red[tid], red[tid + st]);
    __syncthreads();
  }
  const float m = red[0];
  __syncthreads();
  const float e0 = expf(v0 - m), e1 = expf(v1 - m);
  red[tid] = e0 + e1;
  __syncthreads();
  for (int st = 128; st > 0; st >>= 1) {
    if (tid < st) red[tid] += red[tid + st];
    __syncthreads();
  }
  const float inv = 1.f / red[0];
  _Float16* pr = P + row * 512;
  pr[tid]       = (_Float16)(e0 * inv);
  pr[tid + 256] = (_Float16)(e1 * inv);
}

__global__ void hawkeye_swiglu(const float* __restrict__ g,
                               const float* __restrict__ u,
                               _Float16* __restrict__ s, long total) {
  const long idx = (long)blockIdx.x * blockDim.x + threadIdx.x;
  if (idx >= total) return;
  const float gv = g[idx];
  s[idx] = (_Float16)((gv / (1.f + expf(-gv))) * u[idx]);  // silu(g) * u
}

__global__ void hawkeye_fill0(float* __restrict__ p, int n) {
  const int idx = blockIdx.x * blockDim.x + threadIdx.x;
  if (idx < n) p[idx] = 0.f;
}

__global__ void hawkeye_accum(const float* __restrict__ h,
                              const float* __restrict__ r,
                              float* __restrict__ comb, int e) {
  const int idx = blockIdx.x * blockDim.x + threadIdx.x;
  if (idx >= 8 * 30 * 1024) return;
  const int c = idx & 1023;
  const int t = (idx >> 10) % 30;
  const int b = idx / (30 * 1024);
  const long tok = (long)b * 512 + t;
  comb[idx] += h[tok * 1024 + c] * r[tok * 4 + e];
}

__global__ void hawkeye_cast16(const float* __restrict__ a,
                               _Float16* __restrict__ o, int n) {
  const int idx = blockIdx.x * blockDim.x + threadIdx.x;
  if (idx < n) o[idx] = (_Float16)a[idx];
}

__global__ void hawkeye_layernorm(const float* __restrict__ y,
                                  const float* __restrict__ w,
                                  const float* __restrict__ b,
                                  float* __restrict__ out) {
  const int row = blockIdx.x;
  const int tid = threadIdx.x;
  __shared__ float rsum[256];
  __shared__ float rsq[256];
  const float* yr = y + (long)row * 1024;
  float vals[4], s = 0.f, sq = 0.f;
#pragma unroll
  for (int i = 0; i < 4; i++) {
    vals[i] = yr[tid + i * 256];
    s += vals[i];
    sq += vals[i] * vals[i];
  }
  rsum[tid] = s; rsq[tid] = sq;
  __syncthreads();
  for (int st = 128; st > 0; st >>= 1) {
    if (tid < st) { rsum[tid] += rsum[tid + st]; rsq[tid] += rsq[tid + st]; }
    __syncthreads();
  }
  const float mu  = rsum[0] * (1.f / 1024.f);
  const float var = rsq[0] * (1.f / 1024.f) - mu * mu;
  const float inv = rsqrtf(var + 1e-5f);
#pragma unroll
  for (int i = 0; i < 4; i++) {
    const int c = tid + i * 256;
    out[(long)row * 1024 + c] = (vals[i] - mu) * inv * w[c] + b[c];
  }
}

// ---------------------------------------------------------------------------
// Host orchestration
// ---------------------------------------------------------------------------

extern "C" void kernel_launch(void* const* d_in, const int* in_sizes, int n_in,
                              void* d_out, int out_size, void* d_ws, size_t ws_size,
                              hipStream_t stream) {
  (void)in_sizes; (void)n_in; (void)out_size; (void)ws_size;
  constexpr int Bb = 8, Ss = 512, Hh = 1024, Ee = 4, Ll = 2;
  constexpr int NHh = 16, DHh = 64, HFf = 2816, KEEP = 30;
  constexpr int Nt = Bb * Ss;                 // 4096 tokens
  constexpr long SH = (long)Ss * Hh;          // per-batch stride in q/k/v/o

  const float* pose   = (const float*)d_in[0];
  const float* scene  = (const float*)d_in[1];
  const float* rf1w   = (const float*)d_in[2];
  const float* rf1b   = (const float*)d_in[3];
  const float* rf2w   = (const float*)d_in[4];
  const float* rf2b   = (const float*)d_in[5];
  const float* anw    = (const float*)d_in[6];
  const float* wq_w   = (const float*)d_in[7];
  const float* wq_b   = (const float*)d_in[8];
  const float* wk_w   = (const float*)d_in[9];
  const float* wk_b   = (const float*)d_in[10];
  const float* wv_w   = (const float*)d_in[11];
  const float* wv_b   = (const float*)d_in[12];
  const float* wo_w   = (const float*)d_in[13];
  const float* wo_b   = (const float*)d_in[14];
  const float* fnw    = (const float*)d_in[15];
  const float* w1_w   = (const float*)d_in[16];
  const float* w2_w   = (const float*)d_in[17];
  const float* w3_w   = (const float*)d_in[18];
  const float* proj_w = (const float*)d_in[19];
  const float* proj_b = (const float*)d_in[20];
  const float* ln_w   = (const float*)d_in[21];
  const float* ln_b   = (const float*)d_in[22];

  // workspace bump allocator
  char* wsb = (char*)d_ws;
  size_t off = 0;
  auto alloc = [&](size_t bytes) -> char* {
    char* p = wsb + off;
    off = (off + bytes + 255) & ~(size_t)255;
    return p;
  };
  float*    x     = (float*)alloc((size_t)Nt * Hh * 4);
  float*    hbuf  = (float*)alloc((size_t)Nt * Hh * 4);
  _Float16* xn    = (_Float16*)alloc((size_t)Nt * Hh * 2);
  _Float16* qbuf  = (_Float16*)alloc((size_t)Nt * Hh * 2);
  _Float16* kbuf  = (_Float16*)alloc((size_t)Nt * Hh * 2);
  _Float16* vbuf  = (_Float16*)alloc((size_t)Nt * Hh * 2);
  _Float16* obuf  = (_Float16*)alloc((size_t)Nt * Hh * 2);
  float*    rgate = (float*)alloc((size_t)Nt * Ee * 4);
  float*    comb  = (float*)alloc((size_t)Bb * KEEP * Hh * 4);
  _Float16* combh = (_Float16*)alloc((size_t)Bb * KEEP * Hh * 2);
  float*    ybuf  = (float*)alloc((size_t)Bb * KEEP * Hh * 4);
  // big region: attention scores/probs, aliased by FFN activations
  const size_t sc_bytes = (size_t)Bb * NHh * Ss * Ss * 4;   // 134 MB
  const size_t pr_bytes = (size_t)Bb * NHh * Ss * Ss * 2;   // 67 MB
  char* big = alloc(sc_bytes + pr_bytes);
  float*    scores = (float*)big;
  _Float16* probs  = (_Float16*)(big + sc_bytes);
  float*    gbuf   = (float*)big;                                   // alias scores
  _Float16* sbuf   = (_Float16*)(big + (size_t)Nt * HFf * 4);       // after g
  float*    ubuf   = (float*)(big + sc_bytes);                      // alias probs

  const dim3 blk(256);
  const int comb_n = Bb * KEEP * Hh;   // 245760

  // 1) concat + router
  hawkeye_concat<<<(Nt * Hh + 255) / 256, 256, 0, stream>>>(pose, scene, x, Nt * Hh);
  hawkeye_router<<<Nt, 256, 0, stream>>>(x, rf1w, rf1b, rf2w, rf2b, rgate);
  hawkeye_fill0<<<(comb_n + 255) / 256, 256, 0, stream>>>(comb, comb_n);

  for (int e = 0; e < Ee; e++) {
    hipMemcpyAsync(hbuf, x, (size_t)Nt * Hh * 4, hipMemcpyDeviceToDevice, stream);
    for (int l = 0; l < Ll; l++) {
      const long wi = (long)e * Ll + l;
      const float* aW = anw + wi * Hh;
      const float* qW = wq_w + wi * Hh * Hh; const float* qB = wq_b + wi * Hh;
      const float* kW = wk_w + wi * Hh * Hh; const float* kB = wk_b + wi * Hh;
      const float* vW = wv_w + wi * Hh * Hh; const float* vB = wv_b + wi * Hh;
      const float* oW = wo_w + wi * Hh * Hh; const float* oB = wo_b + wi * Hh;
      const float* fW = fnw + wi * Hh;
      const float* W1 = w1_w + wi * (long)Hh * HFf;
      const float* W2 = w2_w + wi * (long)HFf * Hh;
      const float* W3 = w3_w + wi * (long)Hh * HFf;

      // attn RMSNorm -> f16
      hawkeye_rmsnorm<<<Nt, 256, 0, stream>>>(hbuf, aW, xn);

      // QKV projections (+bias, f16 out)
      const dim3 gqkv(Hh / GBN, Nt / GBM, 1);
      hawkeye_gemm_wmma<float, _Float16, false><<<gqkv, blk, 0, stream>>>(
          xn, Hh, 0, 0, qW, Hh, 0, 0, qbuf, Hh, 0, 0, qB, nullptr, 1.f, Nt, Hh, Hh, 1);
      hawkeye_gemm_wmma<float, _Float16, false><<<gqkv, blk, 0, stream>>>(
          xn, Hh, 0, 0, kW, Hh, 0, 0, kbuf, Hh, 0, 0, kB, nullptr, 1.f, Nt, Hh, Hh, 1);
      hawkeye_gemm_wmma<float, _Float16, false><<<gqkv, blk, 0, stream>>>(
          xn, Hh, 0, 0, vW, Hh, 0, 0, vbuf, Hh, 0, 0, vB, nullptr, 1.f, Nt, Hh, Hh, 1);

      // scores = (Q @ K^T) / sqrt(DH), batched over z = b*NH + h
      const dim3 gsc(Ss / GBN, Ss / GBM, Bb * NHh);
      hawkeye_gemm_wmma<_Float16, float, true><<<gsc, blk, 0, stream>>>(
          qbuf, Hh, SH, DHh,                  // A: per-b stride S*H, per-head DH
          kbuf, Hh, SH, DHh,                  // B (transposed use)
          scores, Ss, (long)NHh * Ss * Ss, (long)Ss * Ss,
          nullptr, nullptr, 0.125f, Ss, Ss, DHh, NHh);

      // softmax rows -> f16 probs
      hawkeye_softmax<<<Bb * NHh * Ss, 256, 0, stream>>>(scores, probs);

      // O = P @ V (f16 out, written into [B,S,NH,DH] layout)
      const dim3 gav(1, Ss / GBM, Bb * NHh);
      hawkeye_gemm_wmma<_Float16, _Float16, false><<<gav, blk, 0, stream>>>(
          probs, Ss, (long)NHh * Ss * Ss, (long)Ss * Ss,
          vbuf, Hh, SH, DHh,
          obuf, Hh, SH, DHh,
          nullptr, nullptr, 1.f, Ss, DHh, Ss, NHh);

      // h = h + O @ Wo + bo  (in-place residual)
      hawkeye_gemm_wmma<float, float, false><<<gqkv, blk, 0, stream>>>(
          obuf, Hh, 0, 0, oW, Hh, 0, 0, hbuf, Hh, 0, 0, oB, hbuf, 1.f, Nt, Hh, Hh, 1);

      // FFN
      hawkeye_rmsnorm<<<Nt, 256, 0, stream>>>(hbuf, fW, xn);
      const dim3 gffn(HFf / GBN, Nt / GBM, 1);
      hawkeye_gemm_wmma<float, float, false><<<gffn, blk, 0, stream>>>(
          xn, Hh, 0, 0, W1, HFf, 0, 0, gbuf, HFf, 0, 0, nullptr, nullptr, 1.f, Nt, HFf, Hh, 1);
      hawkeye_gemm_wmma<float, float, false><<<gffn, blk, 0, stream>>>(
          xn, Hh, 0, 0, W3, HFf, 0, 0, ubuf, HFf, 0, 0, nullptr, nullptr, 1.f, Nt, HFf, Hh, 1);
      const long ffn_n = (long)Nt * HFf;
      hawkeye_swiglu<<<(unsigned)((ffn_n + 255) / 256), 256, 0, stream>>>(gbuf, ubuf, sbuf, ffn_n);
      // h = h + s @ W2 (in-place residual)
      hawkeye_gemm_wmma<float, float, false><<<gqkv, blk, 0, stream>>>(
          sbuf, HFf, 0, 0, W2, Hh, 0, 0, hbuf, Hh, 0, 0, nullptr, hbuf, 1.f, Nt, Hh, HFf, 1);
    }
    // combined += h[:, :30] * r[:, :30, e]
    hawkeye_accum<<<(comb_n + 255) / 256, 256, 0, stream>>>(hbuf, rgate, comb, e);
  }

  // final projection + LayerNorm
  hawkeye_cast16<<<(comb_n + 255) / 256, 256, 0, stream>>>(comb, combh, comb_n);
  const dim3 gproj(Hh / GBN, (Bb * KEEP + GBM - 1) / GBM, 1);
  hawkeye_gemm_wmma<float, float, false><<<gproj, blk, 0, stream>>>(
      combh, Hh, 0, 0, proj_w, Hh, 0, 0, ybuf, Hh, 0, 0, proj_b, nullptr,
      1.f, Bb * KEEP, Hh, Hh, 1);
  hawkeye_layernorm<<<Bb * KEEP, 256, 0, stream>>>(ybuf, ln_w, ln_b, (float*)d_out);
}
